// AttentionModule_6674379177997
// MI455X (gfx1250) — matrix operand order repaired and checked by reading, not compiled, not run on hardware
//
#include <hip/hip_runtime.h>
#include <hip/hip_bf16.h>
#include <cstdint>

typedef __attribute__((ext_vector_type(8)))  _Float16 v8h;
typedef __attribute__((ext_vector_type(16))) _Float16 v16h;
typedef __attribute__((ext_vector_type(8)))  float    v8f;

#define B_TOTAL 262144
#define INPUT   66
#define ENODE   50
#define ANODE   20

#define BT   128        // batch rows per workgroup
#define KP1  96         // GEMM1 K: 66 -> 96  (3 x 32)
#define NP1  64         // GEMM1 N: 50 -> 64  (4 x 16)
#define KP2  64         // GEMM2 K: 50 -> 64  (2 x 32)
#define N2   1320       // 66*20
#define NP2  1328       // -> 83 x 16
#define NT2  83

// ---- LDS layout (bytes), dynamic shared ----
#define OFF_W2   0                               // [NP2][KP2] f16 : 169984
#define OFF_E    (OFF_W2 + NP2*KP2*2)            // [BT][NP1]  f16 :  16384
#define OFF_U    (OFF_E  + BT*NP1*2)             // union region   :  84992
#define OFF_XH   OFF_U                           // [BT][KP1]  f16 :  24576 (phase 0/1)
#define OFF_EWH  (OFF_XH + BT*KP1*2)             // [NP1][KP1] f16 :  12288 (phase 0/1)
#define OFF_SC   OFF_U                           // [16][NP2]  f32 :  84992 (phase 2)
#define OFF_EB   (OFF_U  + 16*NP2*4)             // [NP1] f32      :    256
#define OFF_AB   (OFF_EB + NP1*4)                // [66*20] f32    :   5280
#define LDS_BYTES (OFF_AB + INPUT*ANODE*4)       // = 276896 < 320K

// ---- workspace image layout (bytes): pre-packed f16 weights + f32 biases ----
#define WSOFF_W2 0                               // [NP2][KP2] f16 : 169984
#define WSOFF_EW (WSOFF_W2 + NP2*KP2*2)          // [NP1][KP1] f16 :  12288
#define WSOFF_EB (WSOFF_EW + NP1*KP1*2)          // [NP1] f32      :    256
#define WSOFF_AB (WSOFF_EB + NP1*4)              // [66*20] f32    :   5280
#define WS_BYTES (WSOFF_AB + INPUT*ANODE*4)      // = 187808

// A fragment: 16 contiguous f16 at p (K = h*16 .. h*16+15)
static __device__ __forceinline__ v16h ldfrag_a(const _Float16* p) {
  v8h lo = *(const v8h*)p;
  v8h hi = *(const v8h*)(p + 8);
  return __builtin_shufflevector(lo, hi, 0,1,2,3,4,5,6,7,8,9,10,11,12,13,14,15);
}
// B fragment: 16 contiguous f16 at p (K = h*16 .. h*16+15)
static __device__ __forceinline__ v16h ldfrag_b(const _Float16* p) {
  v8h lo = *(const v8h*)p;
  v8h hi = *(const v8h*)(p + 8);
  return __builtin_shufflevector(lo, hi, 0,1,2,3,4,5,6,7,8,9,10,11,12,13,14,15);
}

static __device__ __forceinline__ v8f wmma_f16(v16h a, v16h b, v8f c) {
  return __builtin_amdgcn_wmma_f32_16x16x32_f16(false, a, false, b, (short)0, c, false, false);
}

// Branch-free tanh: native V_TANH_F32 (confirmed in round-2 disasm), else exp-based.
static __device__ __forceinline__ float fast_tanh(float z) {
#if __has_builtin(__builtin_amdgcn_tanhf)
  return __builtin_amdgcn_tanhf(z);
#elif __has_builtin(__builtin_amdgcn_tanh_f32)
  return __builtin_amdgcn_tanh_f32(z);
#else
  float zc = fminf(fmaxf(z, -15.f), 15.f);
  float e  = __expf(2.f * zc);
  return __fdividef(e - 1.f, e + 1.f);
#endif
}

// CDNA5 async copy: global -> LDS, 16B per lane per instruction, ASYNCcnt-tracked.
static __device__ __forceinline__ void async_copy_region(const char* gsrc, char* ldst,
                                                         int nbytes, int tid) {
  unsigned long long gs = (unsigned long long)(uintptr_t)gsrc;
  for (int o = tid * 16; o < nbytes; o += 256 * 16) {
    unsigned int ldsoff = (unsigned int)(uintptr_t)(ldst + o); // low 32 bits = LDS byte offset
    unsigned int voff   = (unsigned int)o;
    asm volatile("global_load_async_to_lds_b128 %0, %1, %2"
                 :: "v"(ldsoff), "v"(voff), "s"(gs) : "memory");
  }
}

// ======== Prep kernel: pack weights once into d_ws (f32 -> f16, transpose) ========
__global__ __launch_bounds__(256)
void prep_kernel(const float* __restrict__ EW, const float* __restrict__ Eb,
                 const float* __restrict__ AW, const float* __restrict__ Ab,
                 char* __restrict__ ws)
{
  _Float16* wW2 = (_Float16*)(ws + WSOFF_W2);
  _Float16* wEW = (_Float16*)(ws + WSOFF_EW);
  float*    wEb = (float*)   (ws + WSOFF_EB);
  float*    wAb = (float*)   (ws + WSOFF_AB);
  const int gt = blockIdx.x * 256 + threadIdx.x;
  const int NT = gridDim.x * 256;

  // W2 N-major: wW2[n*KP2 + k] = AW[i, k, a], n = i*20 + a; zero padded
  for (int e = gt; e < NP2*KP2; e += NT) {
    int n = e >> 6, k = e & 63;
    _Float16 v = (_Float16)0.f;
    if (n < N2 && k < ENODE) {
      int i = n / ANODE, a = n - i*ANODE;
      v = (_Float16)AW[i*(ENODE*ANODE) + k*ANODE + a];
    }
    wW2[e] = v;
  }
  // E_W N-major: wEW[n*KP1 + k] = EW[k][n]; zero padded
  for (int e = gt; e < NP1*KP1; e += NT) {
    int n = e / KP1, k = e - n*KP1;
    wEW[e] = (n < ENODE && k < INPUT) ? (_Float16)EW[k*ENODE + n] : (_Float16)0.f;
  }
  for (int e = gt; e < NP1; e += NT) wEb[e] = (e < ENODE) ? Eb[e] : 0.f;
  for (int e = gt; e < INPUT*ANODE; e += NT) wAb[e] = Ab[e];
}

// ======== Main fused kernel ========
template<bool FROM_WS>
__global__ __launch_bounds__(256)
void attn_fused_kernel(const float* __restrict__ x,  const float* __restrict__ EW,
                       const float* __restrict__ Eb, const float* __restrict__ AW,
                       const float* __restrict__ Ab, const char* __restrict__ ws,
                       float* __restrict__ out)
{
  extern __shared__ char smem[];
  _Float16* sW2 = (_Float16*)(smem + OFF_W2);
  _Float16* sE  = (_Float16*)(smem + OFF_E);
  _Float16* sXh = (_Float16*)(smem + OFF_XH);
  _Float16* sEW = (_Float16*)(smem + OFF_EWH);
  float*    sSc = (float*)   (smem + OFF_SC);
  float*    sEb = (float*)   (smem + OFF_EB);
  float*    sAb = (float*)   (smem + OFF_AB);

  const int tid  = threadIdx.x;
  const int lane = tid & 31;
  const int wv   = tid >> 5;       // wave 0..7
  const int h    = lane >> 4;      // half-wave 0/1
  const int ln16 = lane & 15;
  const int base = blockIdx.x * BT;

  // ======== Phase 0: stage weights (async from ws, or compute) + x tile ========
  if constexpr (FROM_WS) {
    // async DMA of the pre-packed weight image; overlapped with x convert below
    async_copy_region(ws + WSOFF_W2, smem + OFF_W2,  NP2*KP2*2, tid);
    async_copy_region(ws + WSOFF_EW, smem + OFF_EWH, NP1*KP1*2, tid);
    async_copy_region(ws + WSOFF_EB, smem + OFF_EB,  NP1*4 + INPUT*ANODE*4, tid);
  } else {
    for (int n = tid; n < NP2; n += 256) {           // div-free per-row pack
      int i = n / ANODE, a = n - i*ANODE;
      bool valid = (n < N2);
      const float* src = AW + i*(ENODE*ANODE) + a;
      #pragma unroll 4
      for (int k = 0; k < KP2; ++k)
        sW2[n*KP2 + k] = (valid && k < ENODE) ? (_Float16)src[k*ANODE] : (_Float16)0.f;
    }
    for (int e = tid; e < NP1*KP1; e += 256) {
      int n = e / KP1, k = e - n*KP1;
      sEW[e] = (n < ENODE && k < INPUT) ? (_Float16)EW[k*ENODE + n] : (_Float16)0.f;
    }
    for (int e = tid; e < NP1; e += 256) sEb[e] = (e < ENODE) ? Eb[e] : 0.f;
    for (int e = tid; e < INPUT*ANODE; e += 256) sAb[e] = Ab[e];
  }
  // x tile f16, K-padded (needs VALU convert either way)
  for (int e = tid; e < BT*KP1; e += 256) {
    int r = e / KP1, k = e - r*KP1;
    sXh[e] = (k < INPUT) ? (_Float16)x[(size_t)(base + r)*INPUT + k] : (_Float16)0.f;
  }
  if constexpr (FROM_WS) {
    asm volatile("s_wait_asynccnt 0" ::: "memory");
  }
  __syncthreads();

  // ======== Phase 1: E = tanh(x @ E_W + Eb); wave wv owns m-tile wv ========
  {
    const _Float16* arow = sXh + (wv*16 + ln16)*KP1 + h*16;
    v16h a0 = ldfrag_a(arow +  0);
    v16h a1 = ldfrag_a(arow + 32);
    v16h a2 = ldfrag_a(arow + 64);
    #pragma unroll
    for (int nt = 0; nt < 4; ++nt) {
      const _Float16* brow = sEW + (nt*16 + ln16)*KP1 + h*16;
      v16h b0 = ldfrag_b(brow +  0);
      v16h b1 = ldfrag_b(brow + 32);
      v16h b2 = ldfrag_b(brow + 64);
      v8f c = {};
      c = wmma_f16(a0, b0, c);
      c = wmma_f16(a1, b1, c);
      c = wmma_f16(a2, b2, c);
      float bias = sEb[nt*16 + ln16];
      #pragma unroll
      for (int r = 0; r < 8; ++r) {
        float val = fast_tanh(c[r] + bias);
        sE[(wv*16 + r + 8*h)*NP1 + nt*16 + ln16] = (_Float16)val;
      }
    }
  }
  __syncthreads();

  // ======== Phase 2: per m-tile: cooperative scores GEMM + softmax ========
  for (int mt = 0; mt < 8; ++mt) {
    const _Float16* arow = sE + (mt*16 + ln16)*KP2 + h*16;
    v16h a0 = ldfrag_a(arow +  0);
    v16h a1 = ldfrag_a(arow + 32);

    // software-pipelined n-tile loop: preload B for nt+8 before WMMAs for nt,
    // so the per-iteration s_wait_dscnt covers only older loads (no full stall).
    int nt = wv;
    if (nt < NT2) {
      const _Float16* brow = sW2 + (nt*16 + ln16)*KP2 + h*16;
      v16h b0 = ldfrag_b(brow +  0);
      v16h b1 = ldfrag_b(brow + 32);
      while (nt < NT2) {
        const int ntn = nt + 8;
        const int ntc = (ntn < NT2) ? ntn : nt;       // clamped, branch-free preload
        const _Float16* nrow = sW2 + (ntc*16 + ln16)*KP2 + h*16;
        v16h nb0 = ldfrag_b(nrow +  0);
        v16h nb1 = ldfrag_b(nrow + 32);

        v8f c = {};
        c = wmma_f16(a0, b0, c);
        c = wmma_f16(a1, b1, c);
        #pragma unroll
        for (int r = 0; r < 8; ++r)
          sSc[(r + 8*h)*NP2 + nt*16 + ln16] = c[r];

        b0 = nb0; b1 = nb1;
        nt = ntn;
      }
    }
    __syncthreads();

    // softmax over ANODE per (row, i); keep class-1 prob; out = x * p1
    for (int t = tid; t < 16*INPUT; t += 256) {
      int r = t / INPUT, i = t - r*INPUT;
      const float* sc = sSc + r*NP2 + i*ANODE;
      const float* bb = sAb + i*ANODE;
      float s[ANODE];
      float mx = -1e30f;
      #pragma unroll
      for (int a = 0; a < ANODE; ++a) { s[a] = sc[a] + bb[a]; mx = fmaxf(mx, s[a]); }
      float sum = 0.f;
      #pragma unroll
      for (int a = 0; a < ANODE; ++a) sum += __expf(s[a] - mx);
      float p1 = __fdividef(__expf(s[1] - mx), sum);
      size_t gi = (size_t)(base + mt*16 + r)*INPUT + i;
      out[gi] = x[gi] * p1;
    }
    __syncthreads();
  }
}

extern "C" void kernel_launch(void* const* d_in, const int* in_sizes, int n_in,
                              void* d_out, int out_size, void* d_ws, size_t ws_size,
                              hipStream_t stream) {
  (void)in_sizes; (void)n_in; (void)out_size;
  const float* x  = (const float*)d_in[0];
  const float* EW = (const float*)d_in[1];
  const float* Eb = (const float*)d_in[2];
  const float* AW = (const float*)d_in[3];
  const float* Ab = (const float*)d_in[4];
  float* out = (float*)d_out;

  dim3 grid(B_TOTAL / BT);   // 2048 workgroups
  dim3 block(256);           // 8 waves (wave32)

  if (ws_size >= (size_t)WS_BYTES && d_ws != nullptr) {
    prep_kernel<<<dim3(64), block, 0, stream>>>(EW, Eb, AW, Ab, (char*)d_ws);
    attn_fused_kernel<true><<<grid, block, LDS_BYTES, stream>>>(
        x, EW, Eb, AW, Ab, (const char*)d_ws, out);
  } else {
    attn_fused_kernel<false><<<grid, block, LDS_BYTES, stream>>>(
        x, EW, Eb, AW, Ab, nullptr, out);
  }
}
